// MultiheadAttention_67267777790590
// MI455X (gfx1250) — matrix operand order repaired and checked by reading, not compiled, not run on hardware
//
#include <hip/hip_runtime.h>
#include <cstdint>

namespace {

constexpr int kB = 4, kS = 2048, kD = 1024, kH = 8, kDH = 128;
constexpr int kM = kB * kS;  // 8192

typedef __attribute__((ext_vector_type(16))) __bf16 bf16x16;
typedef __attribute__((ext_vector_type(8)))  float  f32x8;
typedef __attribute__((ext_vector_type(4)))  float  f32x4;
typedef __attribute__((ext_vector_type(4)))  unsigned int u32x4;
typedef __attribute__((ext_vector_type(4)))  unsigned short u16x4;
typedef __attribute__((ext_vector_type(4)))  int i32x4;
typedef __attribute__((ext_vector_type(8)))  int i32x8;

typedef __attribute__((may_alias)) f32x4 f32x4_a;
typedef __attribute__((may_alias)) u32x4 u32x4_a;
typedef __attribute__((may_alias)) u16x4 u16x4_a;

union Frag16 { bf16x16 v; u32x4 q[2]; unsigned short s[16]; };

__device__ __forceinline__ unsigned short f2bf(float f) {
  unsigned int u = __builtin_bit_cast(unsigned int, f);
  u += 0x7FFFu + ((u >> 16) & 1u);   // round-to-nearest-even
  return (unsigned short)(u >> 16);
}

__device__ __forceinline__ f32x8 wmma_bf16(bf16x16 a, bf16x16 b, f32x8 c) {
  return __builtin_amdgcn_wmma_f32_16x16x32_bf16(false, a, false, b, (short)0, c, false, false);
}

__device__ __forceinline__ u32x4 lds_b128(const unsigned short* p) {
  return *reinterpret_cast<const u32x4_a*>(p);
}

// A fragment (16 x 32 bf16): lane row = lane&15, half = lane>>4.
__device__ __forceinline__ bf16x16 load_afrag(const unsigned short* base, int stride,
                                              int koff, int lane) {
  const int row = lane & 15, hf = lane >> 4;
  const unsigned short* p = base + row * stride + koff;
  Frag16 f;
  f.q[0] = lds_b128(p + 8 * hf);
  f.q[1] = lds_b128(p + 16 + 8 * hf);
  return f.v;
}

// B fragment (32 x 16 bf16): lane col = lane&15, half = lane>>4.
__device__ __forceinline__ bf16x16 load_bfrag(const unsigned short* base, int stride,
                                              int koff, int lane) {
  const int col = lane & 15, hf = lane >> 4;
  const unsigned short* p = base + col * stride + koff;
  Frag16 f;
  f.q[0] = lds_b128(p + 8 * hf);
  f.q[1] = lds_b128(p + 16 + 8 * hf);
  return f.v;
}

__device__ __forceinline__ float redmax16(float v) {
  v = fmaxf(v, __shfl_xor(v, 1, 32));
  v = fmaxf(v, __shfl_xor(v, 2, 32));
  v = fmaxf(v, __shfl_xor(v, 4, 32));
  v = fmaxf(v, __shfl_xor(v, 8, 32));
  return v;
}
__device__ __forceinline__ float redsum16(float v) {
  v += __shfl_xor(v, 1, 32);
  v += __shfl_xor(v, 2, 32);
  v += __shfl_xor(v, 4, 32);
  v += __shfl_xor(v, 8, 32);
  return v;
}

#define F32X8_ZERO f32x8{0.f,0.f,0.f,0.f,0.f,0.f,0.f,0.f}

// ---------------------------------------------------------------------------
// TDM: 2D tensor_load_to_lds, bf16 (data_size=1 -> 2 bytes), padded LDS rows.
// tile_d0 = contiguous elements per row, tile_d1 = rows,
// stride0 = row stride of the source tensor (elements).
// pad_interval: LDS pad after 2^(pi+1) DWORDs; pad_amount: (pa+1) DWORDs pad.
// ---------------------------------------------------------------------------
__device__ __forceinline__ void tdm_load_2d_bf16(unsigned lds_addr, const unsigned short* gsrc,
                                                 unsigned tile_d0, unsigned tile_d1,
                                                 unsigned stride0, unsigned pad_interval,
                                                 unsigned pad_amount) {
  const unsigned long long ga = (unsigned long long)(uintptr_t)gsrc;
  i32x4 g0;
  g0[0] = 1;                                        // count=1, user descriptor
  g0[1] = (int)lds_addr;                            // lds_addr [63:32]
  g0[2] = (int)(unsigned)(ga & 0xFFFFFFFFu);        // global_addr low
  g0[3] = (int)((unsigned)((ga >> 32) & 0x1FFFFFFu) | (2u << 30));  // addr hi | type=2
  i32x8 g1;
  g1[0] = (int)((1u << 16) | (1u << 20) |           // data_size=2B, pad_enable
                (pad_interval << 22) | (pad_amount << 25));
  g1[1] = (int)(tile_d0 << 16);                     // tensor_dim0 low16 @ [63:48]
  g1[2] = (int)((tile_d0 >> 16) | (tile_d1 << 16)); // tensor_dim0 hi | tensor_dim1 lo
  g1[3] = (int)((tile_d1 >> 16) | (tile_d0 << 16)); // tensor_dim1 hi | tile_dim0
  g1[4] = (int)(tile_d1 & 0xFFFFu);                 // tile_dim1 (tile_dim2=0)
  g1[5] = (int)stride0;                             // tensor_dim0_stride low32
  g1[6] = 0;
  g1[7] = 0;
  asm volatile("tensor_load_to_lds %0, %1" :: "s"(g0), "s"(g1) : "memory");
}

// ---------------------------------------------------------------------------
// Elementwise fp32 -> bf16 (8 elems per thread)
// ---------------------------------------------------------------------------
__global__ __launch_bounds__(256)
void cvt_bf16_kernel(const float* __restrict__ src, unsigned short* __restrict__ dst) {
  const size_t i = ((size_t)blockIdx.x * 256 + threadIdx.x) * 8;
  f32x4 a = *reinterpret_cast<const f32x4_a*>(src + i);
  f32x4 b = *reinterpret_cast<const f32x4_a*>(src + i + 4);
  u16x4 x, y;
#pragma unroll
  for (int e = 0; e < 4; ++e) { x[e] = f2bf(a[e]); y[e] = f2bf(b[e]); }
  *reinterpret_cast<u16x4_a*>(dst + i)     = x;
  *reinterpret_cast<u16x4_a*>(dst + i + 4) = y;
}

// ---------------------------------------------------------------------------
// Y = X @ W^T  with bf16 sources, TDM double-buffered staging, WMMA compute.
// Outputs (any may be null): Yf fp32 row-major; Ybrow bf16 row-major (scaled
// by bscale); Ybt bf16 in head-transposed layout [(b*H+h)*DH+d][token].
// fuse_relu: Yf = residual + relu(acc + bias).
// ---------------------------------------------------------------------------
__global__ __launch_bounds__(256)
void gemm_bf16_tdm_kernel(const unsigned short* __restrict__ Xb,
                          const unsigned short* __restrict__ Wb,
                          float* __restrict__ Yf, unsigned short* __restrict__ Ybrow,
                          unsigned short* __restrict__ Ybt,
                          const float* __restrict__ bias, const float* __restrict__ residual,
                          int M, int N, int K, int fuse_relu, float bscale) {
  constexpr int LSTR = 40;                 // 32 + 8 pad bf16 (80B rows)
  constexpr int TBUF = 128 * LSTR;         // one buffer, elements
  __shared__ __align__(16) unsigned short Alds[2 * TBUF];
  __shared__ __align__(16) unsigned short Blds[2 * TBUF];

  const int tid = threadIdx.x, lane = tid & 31, w = tid >> 5;
  const int m0 = blockIdx.y * 128, n0 = blockIdx.x * 128;
  const int wm = (w >> 1) * 32, wn = (w & 1) * 64;

  f32x8 acc[2][4];
#pragma unroll
  for (int i = 0; i < 2; ++i)
#pragma unroll
    for (int j = 0; j < 4; ++j) acc[i][j] = F32X8_ZERO;

  const int nsteps = K / 32;
  if (w == 0) {  // prologue: stage k-step 0 into buffer 0
    tdm_load_2d_bf16((unsigned)(size_t)&Alds[0], Xb + (size_t)m0 * K, 32, 128, K, 3, 3);
    tdm_load_2d_bf16((unsigned)(size_t)&Blds[0], Wb + (size_t)n0 * K, 32, 128, K, 3, 3);
  }
  int cur = 0;
  for (int it = 0; it < nsteps; ++it) {
    if (w == 0) {
      if (it + 1 < nsteps) {
        const int kn = (it + 1) * 32;
        tdm_load_2d_bf16((unsigned)(size_t)&Alds[(1 - cur) * TBUF],
                         Xb + (size_t)m0 * K + kn, 32, 128, K, 3, 3);
        tdm_load_2d_bf16((unsigned)(size_t)&Blds[(1 - cur) * TBUF],
                         Wb + (size_t)n0 * K + kn, 32, 128, K, 3, 3);
        __builtin_amdgcn_s_wait_tensorcnt((short)2);  // current tile landed
      } else {
        __builtin_amdgcn_s_wait_tensorcnt((short)0);
      }
    }
    __syncthreads();

    const unsigned short* Ab = Alds + cur * TBUF;
    const unsigned short* Bb = Blds + cur * TBUF;
    bf16x16 af[2], bfm[4];
#pragma unroll
    for (int i = 0; i < 2; ++i) af[i] = load_afrag(Ab + (wm + i * 16) * LSTR, LSTR, 0, lane);
#pragma unroll
    for (int j = 0; j < 4; ++j) bfm[j] = load_bfrag(Bb + (wn + j * 16) * LSTR, LSTR, 0, lane);
#pragma unroll
    for (int i = 0; i < 2; ++i)
#pragma unroll
      for (int j = 0; j < 4; ++j)
        acc[i][j] = wmma_bf16(af[i], bfm[j], acc[i][j]);

    __syncthreads();  // protect buffer being refilled next iteration
    cur ^= 1;
  }

  const int hf = lane >> 4, cn = lane & 15;
#pragma unroll
  for (int i = 0; i < 2; ++i)
#pragma unroll
    for (int j = 0; j < 4; ++j)
#pragma unroll
      for (int r = 0; r < 8; ++r) {
        const int row = m0 + wm + i * 16 + r + 8 * hf;
        const int col = n0 + wn + j * 16 + cn;
        const size_t idx = (size_t)row * N + col;
        float vres = acc[i][j][r];
        if (fuse_relu) vres = residual[idx] + fmaxf(vres + bias[col], 0.f);
        if (Yf)    Yf[idx] = vres;
        if (Ybrow) Ybrow[idx] = f2bf(vres * bscale);
        if (Ybt) {
          const int bb = row >> 11, tok = row & 2047;
          const int hh = col >> 7, d = col & 127;
          Ybt[((size_t)(bb * kH + hh) * kDH + d) * kS + tok] = f2bf(vres);
        }
      }
}

// ---------------------------------------------------------------------------
// Flash attention on bf16 qb/kb (row-major [B*S, D]) and vbT ([(b*H+h)*DH+d][tok]).
// qb is pre-scaled by 1/sqrt(D). K and V^T tiles staged by TDM, double-buffered.
// ---------------------------------------------------------------------------
__global__ __launch_bounds__(256)
void attention_kernel(const unsigned short* __restrict__ qb,
                      const unsigned short* __restrict__ kb,
                      const unsigned short* __restrict__ vbT,
                      float* __restrict__ attn) {
  constexpr int KSTR = 136, VSTR = 40, PSTR = 40;
  constexpr int KBUF = 32 * KSTR, VBUF = 128 * VSTR;
  __shared__ __align__(16) unsigned short Klds[2 * KBUF];      // [key][d] padded
  __shared__ __align__(16) unsigned short Vt[2 * VBUF];        // [d][key] padded
  __shared__ __align__(16) unsigned short Plds[8 * 16 * PSTR]; // per-wave P patch

  const int tid = threadIdx.x, lane = tid & 31, w = tid >> 5;
  const int hf = lane >> 4, cn = lane & 15;
  const int qblk = blockIdx.x & 15;
  const int bh = blockIdx.x >> 4;
  const int h = bh & 7, b = bh >> 3;
  const int q0 = qblk * 128 + w * 16;

  unsigned short* Pl = Plds + w * 16 * PSTR;
  const unsigned short* ksrc = kb + (size_t)(b * kS) * kD + h * kDH;
  const unsigned short* vsrc = vbT + (size_t)(b * kH + h) * kDH * kS;

  // Q fragments straight from bf16 global (pre-scaled)
  bf16x16 aq[4];
  {
    const unsigned short* qrow = qb + ((size_t)(b * kS + q0 + cn)) * kD + h * kDH;
#pragma unroll
    for (int kc = 0; kc < 4; ++kc) {
      Frag16 f;
      f.q[0] = *reinterpret_cast<const u32x4_a*>(qrow + kc * 32 + 8 * hf);
      f.q[1] = *reinterpret_cast<const u32x4_a*>(qrow + kc * 32 + 16 + 8 * hf);
      aq[kc] = f.v;
    }
  }

  f32x8 o[8];
#pragma unroll
  for (int t = 0; t < 8; ++t) o[t] = F32X8_ZERO;
  float rowm[8], rowl[8];
#pragma unroll
  for (int r = 0; r < 8; ++r) { rowm[r] = -1e30f; rowl[r] = 0.f; }

  constexpr int nblocks = kS / 32;  // 64
  if (w == 0) {  // prologue: stage key-block 0 into buffer 0
    tdm_load_2d_bf16((unsigned)(size_t)&Klds[0], ksrc, 128, 32, kD, 5, 3);
    tdm_load_2d_bf16((unsigned)(size_t)&Vt[0],   vsrc, 32, 128, kS, 3, 3);
  }
  int cur = 0;
  for (int it = 0; it < nblocks; ++it) {
    if (w == 0) {
      if (it + 1 < nblocks) {
        const int nn = (it + 1) * 32;
        tdm_load_2d_bf16((unsigned)(size_t)&Klds[(1 - cur) * KBUF],
                         ksrc + (size_t)nn * kD, 128, 32, kD, 5, 3);
        tdm_load_2d_bf16((unsigned)(size_t)&Vt[(1 - cur) * VBUF],
                         vsrc + nn, 32, 128, kS, 3, 3);
        __builtin_amdgcn_s_wait_tensorcnt((short)2);
      } else {
        __builtin_amdgcn_s_wait_tensorcnt((short)0);
      }
    }
    __syncthreads();

    const unsigned short* Kb = Klds + cur * KBUF;
    const unsigned short* Vb = Vt + cur * VBUF;

    // S = Q * K^T (contraction over d=128), 2 key tiles of 16
    f32x8 s0 = F32X8_ZERO, s1 = F32X8_ZERO;
#pragma unroll
    for (int kc = 0; kc < 4; ++kc) {
      bf16x16 b0 = load_bfrag(Kb, KSTR, kc * 32, lane);
      bf16x16 b1 = load_bfrag(Kb + 16 * KSTR, KSTR, kc * 32, lane);
      s0 = wmma_bf16(aq[kc], b0, s0);
      s1 = wmma_bf16(aq[kc], b1, s1);
    }

    // Online softmax + P relayout (C -> A) through per-wave LDS patch
    float fsc[8];
#pragma unroll
    for (int r = 0; r < 8; ++r) {
      const float smax = redmax16(fmaxf(s0[r], s1[r]));
      const float mnew = fmaxf(rowm[r], smax);
      const float p0 = __expf(s0[r] - mnew);
      const float p1 = __expf(s1[r] - mnew);
      const float rsum = redsum16(p0 + p1);
      fsc[r] = __expf(rowm[r] - mnew);
      rowl[r] = rowl[r] * fsc[r] + rsum;
      rowm[r] = mnew;
      Pl[(r + 8 * hf) * PSTR + cn]      = f2bf(p0);
      Pl[(r + 8 * hf) * PSTR + cn + 16] = f2bf(p1);
    }
#pragma unroll
    for (int t = 0; t < 8; ++t)
#pragma unroll
      for (int r = 0; r < 8; ++r) o[t][r] *= fsc[r];

    asm volatile("s_wait_dscnt 0" ::: "memory");  // same-wave LDS store->load fence
    bf16x16 pa = load_afrag(Pl, PSTR, 0, lane);
#pragma unroll
    for (int t = 0; t < 8; ++t) {
      bf16x16 vb = load_bfrag(Vb + t * 16 * VSTR, VSTR, 0, lane);
      o[t] = wmma_bf16(pa, vb, o[t]);
    }
    __syncthreads();
    cur ^= 1;
  }

#pragma unroll
  for (int r = 0; r < 8; ++r) {
    const float inv = 1.f / rowl[r];
    const size_t rowbase = ((size_t)(b * kS + q0 + r + 8 * hf)) * kD + h * kDH;
#pragma unroll
    for (int t = 0; t < 8; ++t)
      attn[rowbase + t * 16 + cn] = o[t][r] * inv;
  }
}

// ---------------------------------------------------------------------------
// out = LayerNorm(A + Bp) * g + be   (Bp optional; one block per row, D=1024)
// ---------------------------------------------------------------------------
__global__ __launch_bounds__(256)
void add_ln_kernel(const float* __restrict__ A, const float* __restrict__ Bp,
                   const float* __restrict__ g, const float* __restrict__ be,
                   float* __restrict__ out) {
  __shared__ float sred[16];
  const int row = blockIdx.x, tid = threadIdx.x, lane = tid & 31, w = tid >> 5;
  f32x4 x = *reinterpret_cast<const f32x4_a*>(A + (size_t)row * kD + tid * 4);
  if (Bp) {
    f32x4 y = *reinterpret_cast<const f32x4_a*>(Bp + (size_t)row * kD + tid * 4);
    x += y;
  }
  float s  = x[0] + x[1] + x[2] + x[3];
  float ss = x[0]*x[0] + x[1]*x[1] + x[2]*x[2] + x[3]*x[3];
#pragma unroll
  for (int off = 16; off > 0; off >>= 1) {
    s  += __shfl_xor(s, off, 32);
    ss += __shfl_xor(ss, off, 32);
  }
  if (lane == 0) { sred[w] = s; sred[8 + w] = ss; }
  __syncthreads();
  float S = 0.f, SQ = 0.f;
#pragma unroll
  for (int i = 0; i < 8; ++i) { S += sred[i]; SQ += sred[8 + i]; }
  const float mean = S * (1.f / kD);
  const float var  = SQ * (1.f / kD) - mean * mean;
  const float rs   = rsqrtf(var + 1e-5f);
#pragma unroll
  for (int e = 0; e < 4; ++e) {
    const int c = tid * 4 + e;
    out[(size_t)row * kD + c] = (x[e] - mean) * rs * g[c] + be[c];
  }
}

}  // namespace

extern "C" void kernel_launch(void* const* d_in, const int* in_sizes, int n_in,
                              void* d_out, int out_size, void* d_ws, size_t ws_size,
                              hipStream_t stream) {
  (void)in_sizes; (void)n_in; (void)out_size; (void)ws_size;
  const float* q    = (const float*)d_in[0];
  const float* k    = (const float*)d_in[1];
  const float* v    = (const float*)d_in[2];
  const float* Wq   = (const float*)d_in[3];
  const float* Wk   = (const float*)d_in[4];
  const float* Wv   = (const float*)d_in[5];
  const float* Wo   = (const float*)d_in[6];
  const float* bo   = (const float*)d_in[7];
  const float* ln1g = (const float*)d_in[8];
  const float* ln1b = (const float*)d_in[9];
  const float* ln2g = (const float*)d_in[10];
  const float* ln2b = (const float*)d_in[11];
  float* out = (float*)d_out;

  const size_t E = (size_t)kM * kD;   // 8388608
  const size_t WE = (size_t)kD * kD;  // 1048576

  // Workspace layout with lifetime-based overlap (total ~143 MB):
  float* qp = (float*)d_ws;                               // [E] fp32, live -> LN1
  unsigned short* cq  = (unsigned short*)(qp + E);        // [E] bf16 input copies
  unsigned short* ck  = cq + E;
  unsigned short* cv  = ck + E;
  unsigned short* qb  = cv + E;                           // [E] bf16 qp (scaled)
  unsigned short* kb  = qb + E;                           // [E] bf16 kp
  unsigned short* vbT = kb + E;                           // [E] bf16 vp, transposed
  unsigned short* Wqb = vbT + E;                          // [WE] x4 bf16 weights
  unsigned short* Wkb = Wqb + WE;
  unsigned short* Wvb = Wkb + WE;
  unsigned short* Wob = Wvb + WE;
  float* attn = (float*)cq;        // reuse cq+ck (dead after projections)
  float* t    = (float*)cv;        // reuse cv+qb (dead after attention)
  unsigned short* tb = kb;         // reuse kb   (dead after attention)

  const int cvtE = (int)(E / 2048), cvtW = (int)(WE / 2048);
  cvt_bf16_kernel<<<cvtE, 256, 0, stream>>>(q, cq);
  cvt_bf16_kernel<<<cvtE, 256, 0, stream>>>(k, ck);
  cvt_bf16_kernel<<<cvtE, 256, 0, stream>>>(v, cv);
  cvt_bf16_kernel<<<cvtW, 256, 0, stream>>>(Wq, Wqb);
  cvt_bf16_kernel<<<cvtW, 256, 0, stream>>>(Wk, Wkb);
  cvt_bf16_kernel<<<cvtW, 256, 0, stream>>>(Wv, Wvb);
  cvt_bf16_kernel<<<cvtW, 256, 0, stream>>>(Wo, Wob);

  dim3 gg(kD / 128, kM / 128);  // (8, 64)
  // qp fp32 + qb bf16 (pre-scaled by 1/sqrt(D)=1/32)
  gemm_bf16_tdm_kernel<<<gg, 256, 0, stream>>>(cq, Wqb, qp, qb, nullptr,
                                               nullptr, nullptr, kM, kD, kD, 0, 0.03125f);
  // kb bf16 row-major only
  gemm_bf16_tdm_kernel<<<gg, 256, 0, stream>>>(ck, Wkb, nullptr, kb, nullptr,
                                               nullptr, nullptr, kM, kD, kD, 0, 1.0f);
  // vbT bf16 head-transposed only
  gemm_bf16_tdm_kernel<<<gg, 256, 0, stream>>>(cv, Wvb, nullptr, nullptr, vbT,
                                               nullptr, nullptr, kM, kD, kD, 0, 1.0f);
  // attention
  attention_kernel<<<kB * kH * (kS / 128), 256, 0, stream>>>(qb, kb, vbT, attn);
  // t = LN(qp + attn)
  add_ln_kernel<<<kM, 256, 0, stream>>>(qp, attn, ln1g, ln1b, t);
  // tb = bf16(t)
  cvt_bf16_kernel<<<cvtE, 256, 0, stream>>>(t, tb);
  // out = t + relu(t @ Wo^T + bo)
  gemm_bf16_tdm_kernel<<<gg, 256, 0, stream>>>(tb, Wob, out, nullptr, nullptr,
                                               bo, t, kM, kD, kD, 1, 1.0f);
  // out = LN(out), in place
  add_ln_kernel<<<kM, 256, 0, stream>>>(out, nullptr, ln2g, ln2b, out);
}
